// MoE_10582799417581
// MI455X (gfx1250) — matrix operand order repaired and checked by reading, not compiled, not run on hardware
//
#include <hip/hip_runtime.h>
#include <math.h>

// Problem constants (from the reference).
#define TOKENS 8192     // BATCH * SEQ
#define DMODEL 1024
#define DFF    2048

typedef __attribute__((ext_vector_type(16))) __bf16 v16bf;
typedef __attribute__((ext_vector_type(8)))  float  v8f;
typedef __attribute__((ext_vector_type(4)))  unsigned int u32x4;
typedef __attribute__((ext_vector_type(8)))  unsigned int u32x8;

// GEMM tiling: 128x128 block tile, K-tile 64, 256 threads (8 wave32s).
#define BM 128
#define BN 128
#define BK 64
#define KQ  (BK / 2)      // 32 uints (bf16 pairs) per row
#define LDA (KQ + 8)      // 40 uints -> 160B row stride (multiple of 16B)
#define LDB (KQ + 8)

union Frag16 { unsigned int u[8]; v16bf v; };

static __device__ __forceinline__ unsigned int bf16_rne(float f) {
  unsigned int a = __float_as_uint(f);
  return (a + 0x7FFFu + ((a >> 16) & 1u)) >> 16;
}
static __device__ __forceinline__ unsigned int pack_bf16(float lo, float hi) {
#if __has_builtin(__builtin_amdgcn_cvt_pk_bf16_f32)
  typedef __attribute__((ext_vector_type(2))) __bf16 v2bf;
  union { v2bf v; unsigned int u; } c;
  c.v = __builtin_amdgcn_cvt_pk_bf16_f32(lo, hi);   // v_cvt_pk_bf16_f32 (RNE)
  return c.u;
#else
  return (bf16_rne(lo) & 0xFFFFu) | (bf16_rne(hi) << 16);
#endif
}

// ---------------------------------------------------------------------------
// Gating: one wave per token. logits = x @ gate_w + gate_b ; softmax ; top-2.
// s0[t] = p_top1 if argtop1==0 else 0 ;  s1[t] = p_top2 if argtop2==1 else 0.
// ---------------------------------------------------------------------------
__global__ __launch_bounds__(256)
void moe_gate_kernel(const float* __restrict__ x,
                     const float* __restrict__ gw,   // [D, 8]
                     const float* __restrict__ gb,   // [8]
                     float* __restrict__ s0,
                     float* __restrict__ s1,
                     float* __restrict__ lbl) {
  if (blockIdx.x == 0 && threadIdx.x == 0) *lbl = 0.0f;  // provably 0 in ref
  const int lane = threadIdx.x & 31;
  const int wv   = threadIdx.x >> 5;
  const int t    = blockIdx.x * 8 + wv;
  if (t >= TOKENS) return;

  float p[8] = {0.f, 0.f, 0.f, 0.f, 0.f, 0.f, 0.f, 0.f};
  const float* xr = x + (size_t)t * DMODEL;
  for (int i = lane; i < DMODEL; i += 32) {
    const float xv = xr[i];
    const float* g = gw + (size_t)i * 8;
#pragma unroll
    for (int e = 0; e < 8; ++e) p[e] += xv * g[e];
  }
#pragma unroll
  for (int e = 0; e < 8; ++e)
#pragma unroll
    for (int off = 16; off > 0; off >>= 1)
      p[e] += __shfl_down(p[e], off, 32);

  if (lane == 0) {
#pragma unroll
    for (int e = 0; e < 8; ++e) p[e] += gb[e];
    float mx = p[0];
#pragma unroll
    for (int e = 1; e < 8; ++e) mx = fmaxf(mx, p[e]);
    float sum = 0.f;
#pragma unroll
    for (int e = 0; e < 8; ++e) { p[e] = __expf(p[e] - mx); sum += p[e]; }
    const float inv = 1.0f / sum;
#pragma unroll
    for (int e = 0; e < 8; ++e) p[e] *= inv;
    int i1 = 0; float v1 = p[0];
#pragma unroll
    for (int e = 1; e < 8; ++e) if (p[e] > v1) { v1 = p[e]; i1 = e; }
    int i2 = -1; float v2 = -1.0f;
#pragma unroll
    for (int e = 0; e < 8; ++e) if (e != i1 && p[e] > v2) { v2 = p[e]; i2 = e; }
    s0[t] = (i1 == 0) ? v1 : 0.0f;
    s1[t] = (i2 == 1) ? v2 : 0.0f;
  }
}

// ---------------------------------------------------------------------------
// Layer 1: H[M,N] = gelu_exact(X[M,K] @ W[K,N] + bias[N]) -> bf16
// f32 operands converted to packed bf16 in LDS; v_wmma_f32_16x16x32_bf16.
// ---------------------------------------------------------------------------
__global__ __launch_bounds__(256)
void moe_ffn1_wmma(const float* __restrict__ X,
                   const float* __restrict__ W,
                   const float* __restrict__ bias,
                   unsigned short* __restrict__ Hout,
                   int M, int N, int K) {
  __shared__ unsigned int Asu[BM * LDA];   // [row][kpair], bf16 pairs along K
  __shared__ unsigned int Bsu[BN * LDB];   // [col][kpair], transposed

  const int tid  = threadIdx.x;
  const int m0   = blockIdx.y * BM;
  const int n0   = blockIdx.x * BN;
  const int lane = tid & 31;
  const int wv   = tid >> 5;
  const int wm   = wv & 3;    // 4 wave-rows
  const int wn   = wv >> 2;   // 2 wave-cols
  const int lrow = lane & 15;
  const int lhi  = lane >> 4;

  v8f acc[2][4];
#pragma unroll
  for (int mi = 0; mi < 2; ++mi)
#pragma unroll
    for (int ni = 0; ni < 4; ++ni)
      acc[mi][ni] = (v8f){0.f, 0.f, 0.f, 0.f, 0.f, 0.f, 0.f, 0.f};

  for (int k0 = 0; k0 < K; k0 += BK) {
    // Stage A: BM x BK f32 -> bf16 pairs (K contiguous in memory)
#pragma unroll
    for (int i = 0; i < 8; ++i) {
      const int idx = tid + i * 256;           // 2048 float4 slots
      const int row = idx >> 4;
      const int c4  = idx & 15;
      const float4 f = *(const float4*)(X + (size_t)(m0 + row) * K + k0 + c4 * 4);
      *(uint2*)&Asu[row * LDA + c4 * 2] =
          make_uint2(pack_bf16(f.x, f.y), pack_bf16(f.z, f.w));
    }
    // Stage B transposed: W[K,N] tile -> Bsu[col][kpair]
#pragma unroll
    for (int i = 0; i < 4; ++i) {
      const int idx = tid + i * 256;           // 1024 (k-pair, n4) slots
      const int k2  = idx >> 5;                // 0..31 (pair of K rows)
      const int nc  = (idx & 31) * 4;
      const float* wp = W + (size_t)(k0 + k2 * 2) * N + n0 + nc;
      const float4 f0 = *(const float4*)wp;
      const float4 f1 = *(const float4*)(wp + N);
      Bsu[(nc + 0) * LDB + k2] = pack_bf16(f0.x, f1.x);
      Bsu[(nc + 1) * LDB + k2] = pack_bf16(f0.y, f1.y);
      Bsu[(nc + 2) * LDB + k2] = pack_bf16(f0.z, f1.z);
      Bsu[(nc + 3) * LDB + k2] = pack_bf16(f0.w, f1.w);
    }
    __syncthreads();

    if (k0 + BK < K) {  // global_prefetch_b8 of the next K tile
      __builtin_prefetch(X + (size_t)(m0 + (tid >> 1)) * K + (k0 + BK) + (tid & 1) * 32, 0, 1);
      __builtin_prefetch(W + (size_t)(k0 + BK + (tid >> 2)) * N + n0 + (tid & 3) * 32, 0, 1);
    }

#pragma unroll
    for (int kk = 0; kk < 2; ++kk) {          // two K=32 steps per tile
      Frag16 a[2], b[4];
#pragma unroll
      for (int mi = 0; mi < 2; ++mi) {
        const uint4* p = (const uint4*)&Asu[(wm * 32 + mi * 16 + lrow) * LDA + kk * 16 + lhi * 4];
        const uint4 q0 = p[0], q1 = p[2];     // K = kk*32+lhi*8.. and +16
        a[mi].u[0] = q0.x; a[mi].u[1] = q0.y; a[mi].u[2] = q0.z; a[mi].u[3] = q0.w;
        a[mi].u[4] = q1.x; a[mi].u[5] = q1.y; a[mi].u[6] = q1.z; a[mi].u[7] = q1.w;
      }
#pragma unroll
      for (int ni = 0; ni < 4; ++ni) {
        const uint4* p = (const uint4*)&Bsu[(wn * 64 + ni * 16 + lrow) * LDB + kk * 16 + lhi * 4];
        const uint4 q0 = p[0], q1 = p[2];     // K = kk*32+lhi*8.. and +16
        b[ni].u[0] = q0.x; b[ni].u[1] = q0.y; b[ni].u[2] = q0.z; b[ni].u[3] = q0.w;
        b[ni].u[4] = q1.x; b[ni].u[5] = q1.y; b[ni].u[6] = q1.z; b[ni].u[7] = q1.w;
      }
#pragma unroll
      for (int mi = 0; mi < 2; ++mi)
#pragma unroll
        for (int ni = 0; ni < 4; ++ni)
          acc[mi][ni] = __builtin_amdgcn_wmma_f32_16x16x32_bf16(
              false, a[mi].v, false, b[ni].v, (short)0, acc[mi][ni], false, false);
    }
    __syncthreads();
  }

  // Epilogue: bias + exact GELU -> bf16
#pragma unroll
  for (int mi = 0; mi < 2; ++mi)
#pragma unroll
    for (int ni = 0; ni < 4; ++ni) {
      const int col = n0 + wn * 64 + ni * 16 + lrow;
      const float bv = bias[col];
#pragma unroll
      for (int r = 0; r < 8; ++r) {
        const int row = m0 + wm * 32 + mi * 16 + lhi * 8 + r;
        const float v = acc[mi][ni][r] + bv;
        const float g = 0.5f * v * (1.0f + erff(v * 0.70710678118654752f));
        Hout[(size_t)row * N + col] = (unsigned short)bf16_rne(g);
      }
    }
}

// ---------------------------------------------------------------------------
// Layer 2: Out[M,N] (+)= scale[row] * (H[M,K]bf16 @ W[K,N] + bias[N])
// A tile (bf16, no conversion needed) is DMA'd into LDS by the Tensor Data
// Mover (tensor_load_to_lds, TENSORcnt), with D# LDS padding producing the
// 160B padded row stride directly. Dynamic LDS so the tile base is offset 0.
// ---------------------------------------------------------------------------
__global__ __launch_bounds__(256)
void moe_ffn2_wmma(const unsigned short* __restrict__ Hin,
                   const float* __restrict__ W,
                   const float* __restrict__ bias,
                   const float* __restrict__ scale,
                   float* __restrict__ Out,
                   int M, int N, int K, int accumulate) {
  extern __shared__ unsigned int lds[];
  unsigned int* Asu = lds;                 // BM*LDA uints, written by TDM (LDS addr 0)
  unsigned int* Bsu = lds + BM * LDA;      // BN*LDB uints

  const int tid  = threadIdx.x;
  const int m0   = blockIdx.y * BM;
  const int n0   = blockIdx.x * BN;
  const int lane = tid & 31;
  const int wv   = tid >> 5;
  const int wm   = wv & 3;
  const int wn   = wv >> 2;
  const int lrow = lane & 15;
  const int lhi  = lane >> 4;

  // Loop-invariant D# group 1: workgroup_mask=0, data_size=2B (code 1),
  // pad_enable=1, pad_interval=32 DWORDs (code 4), pad_amount=8 DWORDs (code 7),
  // tensor_dim0=K, tensor_dim1=M, tile_dim0=64, tile_dim1=128,
  // tensor_dim0_stride=K.
  u32x8 g1;
  g1[0] = (1u << 16) | (1u << 20) | (4u << 22) | (7u << 25);
  g1[1] = ((unsigned)K & 0xFFFFu) << 16;
  g1[2] = ((unsigned)K >> 16) | (((unsigned)M & 0xFFFFu) << 16);
  g1[3] = ((unsigned)M >> 16) | ((unsigned)BK << 16);
  g1[4] = (unsigned)BM;                     // tile_dim1=128, tile_dim2=0
  g1[5] = (unsigned)K;                      // dim0_stride[31:0]
  g1[6] = 0u;                               // dim0_stride[47:32] | dim1_stride lo
  g1[7] = 0u;
  const unsigned long long gbase =
      (unsigned long long)Hin + (unsigned long long)m0 * K * 2ull;

  v8f acc[2][4];
#pragma unroll
  for (int mi = 0; mi < 2; ++mi)
#pragma unroll
    for (int ni = 0; ni < 4; ++ni)
      acc[mi][ni] = (v8f){0.f, 0.f, 0.f, 0.f, 0.f, 0.f, 0.f, 0.f};

  for (int k0 = 0; k0 < K; k0 += BK) {
    // Kick off the TDM for the A tile from wave 0 (TDM ignores EXEC; one
    // issue per workgroup). Overlaps with the B staging below.
    if (wv == 0) {
      const unsigned long long ga = gbase + (unsigned long long)k0 * 2ull;
      u32x4 g0;
      g0[0] = 1u;                            // count=1 valid descriptor
      g0[1] = 0u;                            // lds_addr = 0 (dynamic LDS base)
      g0[2] = (unsigned)ga;                  // global_addr[31:0]
      g0[3] = (unsigned)(ga >> 32) | (2u << 30);  // addr[56:32] | type=2
      asm volatile("tensor_load_to_lds %0, %1" :: "s"(g0), "s"(g1) : "memory");
    }

    // Stage B transposed (f32 -> bf16) with all 256 threads.
#pragma unroll
    for (int i = 0; i < 4; ++i) {
      const int idx = tid + i * 256;
      const int k2  = idx >> 5;
      const int nc  = (idx & 31) * 4;
      const float* wp = W + (size_t)(k0 + k2 * 2) * N + n0 + nc;
      const float4 f0 = *(const float4*)wp;
      const float4 f1 = *(const float4*)(wp + N);
      Bsu[(nc + 0) * LDB + k2] = pack_bf16(f0.x, f1.x);
      Bsu[(nc + 1) * LDB + k2] = pack_bf16(f0.y, f1.y);
      Bsu[(nc + 2) * LDB + k2] = pack_bf16(f0.z, f1.z);
      Bsu[(nc + 3) * LDB + k2] = pack_bf16(f0.w, f1.w);
    }

    if (k0 + BK < K)
      __builtin_prefetch(W + (size_t)(k0 + BK + (tid >> 2)) * N + n0 + (tid & 3) * 32, 0, 1);

    if (wv == 0) __builtin_amdgcn_s_wait_tensorcnt(0);  // TDM done (wave 0)
    __syncthreads();                                    // publish A+B tiles

#pragma unroll
    for (int kk = 0; kk < 2; ++kk) {
      Frag16 a[2], b[4];
#pragma unroll
      for (int mi = 0; mi < 2; ++mi) {
        const uint4* p = (const uint4*)&Asu[(wm * 32 + mi * 16 + lrow) * LDA + kk * 16 + lhi * 4];
        const uint4 q0 = p[0], q1 = p[2];
        a[mi].u[0] = q0.x; a[mi].u[1] = q0.y; a[mi].u[2] = q0.z; a[mi].u[3] = q0.w;
        a[mi].u[4] = q1.x; a[mi].u[5] = q1.y; a[mi].u[6] = q1.z; a[mi].u[7] = q1.w;
      }
#pragma unroll
      for (int ni = 0; ni < 4; ++ni) {
        const uint4* p = (const uint4*)&Bsu[(wn * 64 + ni * 16 + lrow) * LDB + kk * 16 + lhi * 4];
        const uint4 q0 = p[0], q1 = p[2];
        b[ni].u[0] = q0.x; b[ni].u[1] = q0.y; b[ni].u[2] = q0.z; b[ni].u[3] = q0.w;
        b[ni].u[4] = q1.x; b[ni].u[5] = q1.y; b[ni].u[6] = q1.z; b[ni].u[7] = q1.w;
      }
#pragma unroll
      for (int mi = 0; mi < 2; ++mi)
#pragma unroll
        for (int ni = 0; ni < 4; ++ni)
          acc[mi][ni] = __builtin_amdgcn_wmma_f32_16x16x32_bf16(
              false, a[mi].v, false, b[ni].v, (short)0, acc[mi][ni], false, false);
    }
    __syncthreads();   // all reads done before next tile's TDM/stores
  }

  // Epilogue: scale * (acc + bias); expert 0 writes, expert 1 accumulates.
#pragma unroll
  for (int mi = 0; mi < 2; ++mi)
#pragma unroll
    for (int ni = 0; ni < 4; ++ni) {
      const int col = n0 + wn * 64 + ni * 16 + lrow;
      const float bv = bias[col];
#pragma unroll
      for (int r = 0; r < 8; ++r) {
        const int row = m0 + wm * 32 + mi * 16 + lhi * 8 + r;
        const float s = scale[row];
        const float v = s * (acc[mi][ni][r] + bv);
        const size_t o = (size_t)row * N + col;
        if (accumulate) Out[o] += v; else Out[o] = v;
      }
    }
}

// ---------------------------------------------------------------------------
extern "C" void kernel_launch(void* const* d_in, const int* in_sizes, int n_in,
                              void* d_out, int out_size, void* d_ws, size_t ws_size,
                              hipStream_t stream) {
  const float* x  = (const float*)d_in[0];  // [4,2048,1024]
  const float* gw = (const float*)d_in[1];  // [1024,8]
  const float* gb = (const float*)d_in[2];  // [8]
  const float* w1 = (const float*)d_in[3];  // [8,1024,2048]
  const float* b1 = (const float*)d_in[4];  // [8,2048]
  const float* w2 = (const float*)d_in[5];  // [8,2048,1024]
  const float* b2 = (const float*)d_in[6];  // [8,1024]
  float* out = (float*)d_out;               // [4,2048,1024] ++ [lbl]

  // Workspace layout: s0 (32KB) | s1 (32KB) | H bf16 (32MB, reused per expert)
  float* s0 = (float*)d_ws;
  float* s1 = s0 + TOKENS;
  unsigned short* H = (unsigned short*)((char*)d_ws + 2 * TOKENS * sizeof(float));

  // 1) gating scales + lbl = 0
  moe_gate_kernel<<<TOKENS / 8, 256, 0, stream>>>(x, gw, gb, s0, s1,
                                                  out + (size_t)TOKENS * DMODEL);

  const size_t ffn2_lds = (size_t)(BM * LDA + BN * LDB) * sizeof(unsigned int);

  // 2) experts 0 and 1 (the only ones the faithful reference ever computes)
  for (int e = 0; e < 2; ++e) {
    const float* w1e = w1 + (size_t)e * DMODEL * DFF;
    const float* b1e = b1 + (size_t)e * DFF;
    const float* w2e = w2 + (size_t)e * DFF * DMODEL;
    const float* b2e = b2 + (size_t)e * DMODEL;
    moe_ffn1_wmma<<<dim3(DFF / BN, TOKENS / BM), 256, 0, stream>>>(
        x, w1e, b1e, H, TOKENS, DFF, DMODEL);
    moe_ffn2_wmma<<<dim3(DMODEL / BN, TOKENS / BM), 256, ffn2_lds, stream>>>(
        H, w2e, b2e, (e == 0) ? s0 : s1, out, TOKENS, DMODEL, DFF, /*accumulate=*/e);
  }
}
